// Model_35991825940825
// MI455X (gfx1250) — compile-verified, matrix operand order checked by reading, not run on hardware
//
#include <hip/hip_runtime.h>
#include <hip/hip_bf16.h>

// Problem constants (from reference)
#define VOCAB  50000
#define EMBED  128
#define HIDDEN 128
#define SEQ    512
#define BATCH  256

typedef __attribute__((ext_vector_type(16))) __bf16 v16bf;
typedef __attribute__((ext_vector_type(8)))  __bf16 bf8v;
typedef __attribute__((ext_vector_type(8)))  float  v8f;
typedef __attribute__((ext_vector_type(4)))  float  f4v;

__device__ __forceinline__ float sigmoid_f(float x) {
    return 1.0f / (1.0f + __expf(-x));
}

__device__ __forceinline__ unsigned short f32_to_bf16_bits(float x) {
    union { float f; unsigned u; } t; t.f = x;
    unsigned r = t.u + 0x7FFFu + ((t.u >> 16) & 1u);   // round-to-nearest-even
    return (unsigned short)(r >> 16);
}

// ---------------------------------------------------------------------------
// Pre-pass: transpose fp32 weight [K=128][N=512] -> bf16 [N=512][K=128]
// so WMMA B-tiles (column of W = row of W^T) load as contiguous bf16 chunks.
// ---------------------------------------------------------------------------
__global__ void transpose_w_bf16(const float* __restrict__ src,
                                 unsigned short* __restrict__ dst,
                                 int K, int N) {
    int i = blockIdx.x * blockDim.x + threadIdx.x;
    if (i >= K * N) return;
    int n = i / K;            // dst row   (column of src)
    int k = i % K;            // dst col   (row of src)
    dst[i] = f32_to_bf16_bits(src[k * N + n]);
}

// h0 (bf16) = 0, c (f32) = 0
__global__ void init_state(unsigned short* __restrict__ h0,
                           float* __restrict__ c, int n) {
    int i = blockIdx.x * blockDim.x + threadIdx.x;
    if (i < n) { h0[i] = 0; c[i] = 0.0f; }
}

// softmax over a size-1 axis is identically 1.0
__global__ void write_out(float* __restrict__ out, int n) {
    int i = blockIdx.x * blockDim.x + threadIdx.x;
    if (i < n) out[i] = 1.0f;
}

// ---------------------------------------------------------------------------
// One LSTM timestep. One wave (32 lanes) per block.
//   block (bx, by) owns batch rows [bx*16, bx*16+16) x hidden cols [by*16, +16)
//   and computes all 4 gates (i,f,g,o) for that tile with
//   v_wmma_f32_16x16x32_bf16, K = 128 (embed) + 128 (recurrent).
// A-tile (16x32 bf16) per-lane layout: lane<16 -> row M=lane, K = {k0..k0+7,
// k0+16..k0+23}; lane>=16 same row set with k0 += 8. B mirrors A with N in
// place of M, which matches contiguous rows of the transposed weight.
// C/D (16x16 f32): lane<16 -> (M=j, N=lane); lane>=16 -> (M=j+8, N=lane-16).
// ---------------------------------------------------------------------------
__global__ void __launch_bounds__(32)
lstm_step(const int*            __restrict__ texts,   // [BATCH][SEQ]
          const float*          __restrict__ emb,     // [VOCAB][EMBED]
          const unsigned short* __restrict__ WkT,     // bf16 [4H][EMBED]
          const unsigned short* __restrict__ WrT,     // bf16 [4H][HIDDEN]
          const float*          __restrict__ bias,    // [4H]
          const unsigned short* __restrict__ h_in,    // bf16 [BATCH][HIDDEN]
          unsigned short*       __restrict__ h_out,   // bf16 [BATCH][HIDDEN]
          float*                __restrict__ c,       // f32  [BATCH][HIDDEN]
          int t) {
    const int lane = threadIdx.x;
    const int lrow = lane & 15;
    const int half = lane >> 4;
    const int m0   = blockIdx.x * 16;     // batch tile base
    const int n0   = blockIdx.y * 16;     // hidden tile base
    const int m    = m0 + lrow;           // A row (batch) for this lane
    const int khalf = half * 8;

    // Accumulators, one per gate, initialized with the (broadcast) bias.
    v8f acc[4];
#pragma unroll
    for (int g = 0; g < 4; ++g) {
        float bg = bias[g * HIDDEN + n0 + lrow];
#pragma unroll
        for (int j = 0; j < 8; ++j) acc[g][j] = bg;
    }

    // ---- input projection: A from embedding row (fp32 -> bf16 on the fly)
    const int tok = texts[m * SEQ + t];
    const float* erow = emb + (long)tok * EMBED;
#pragma unroll
    for (int kb = 0; kb < 4; ++kb) {
        const int k0 = kb * 32 + khalf;
        v16bf a;
#pragma unroll
        for (int j = 0; j < 8; ++j) {
            a[j]     = (__bf16)erow[k0 + j];
            a[j + 8] = (__bf16)erow[k0 + 16 + j];
        }
#pragma unroll
        for (int g = 0; g < 4; ++g) {
            const __bf16* wrow =
                (const __bf16*)(WkT + (long)(g * HIDDEN + n0 + lrow) * EMBED);
            union { v16bf v; bf8v h2[2]; } bu;
            bu.h2[0] = *(const bf8v*)(wrow + k0);
            bu.h2[1] = *(const bf8v*)(wrow + k0 + 16);
            acc[g] = __builtin_amdgcn_wmma_f32_16x16x32_bf16(
                false, a, false, bu.v, (short)0, acc[g], false, false);
        }
    }

    // ---- recurrent projection: A from previous h (already bf16)
    const __bf16* hrow = (const __bf16*)(h_in + m * HIDDEN);
#pragma unroll
    for (int kb = 0; kb < 4; ++kb) {
        const int k0 = kb * 32 + khalf;
        union { v16bf v; bf8v h2[2]; } au;
        au.h2[0] = *(const bf8v*)(hrow + k0);
        au.h2[1] = *(const bf8v*)(hrow + k0 + 16);
#pragma unroll
        for (int g = 0; g < 4; ++g) {
            const __bf16* wrow =
                (const __bf16*)(WrT + (long)(g * HIDDEN + n0 + lrow) * HIDDEN);
            union { v16bf v; bf8v h2[2]; } bu;
            bu.h2[0] = *(const bf8v*)(wrow + k0);
            bu.h2[1] = *(const bf8v*)(wrow + k0 + 16);
            acc[g] = __builtin_amdgcn_wmma_f32_16x16x32_bf16(
                false, au.v, false, bu.v, (short)0, acc[g], false, false);
        }
    }

    // ---- fused gate epilogue: c/h update for this tile (tile-private in c,
    //      double-buffered in h so other tiles' reads of h_in are race-free)
    const int ncol  = n0 + lrow;
    const int mbase = m0 + half * 8;
#pragma unroll
    for (int j = 0; j < 8; ++j) {
        const int idx = (mbase + j) * HIDDEN + ncol;
        float iv = sigmoid_f(acc[0][j]);
        float fv = sigmoid_f(acc[1][j]);
        float gv = tanhf(acc[2][j]);
        float ov = sigmoid_f(acc[3][j]);
        float cn = fv * c[idx] + iv * gv;
        c[idx] = cn;
        h_out[idx] = f32_to_bf16_bits(ov * tanhf(cn));
    }
}

// ---------------------------------------------------------------------------
extern "C" void kernel_launch(void* const* d_in, const int* in_sizes, int n_in,
                              void* d_out, int out_size, void* d_ws, size_t ws_size,
                              hipStream_t stream) {
    const int*   texts = (const int*)  d_in[0];
    const float* emb   = (const float*)d_in[1];
    const float* Wk    = (const float*)d_in[2];   // [EMBED][4H]
    const float* Wr    = (const float*)d_in[3];   // [HIDDEN][4H]
    const float* bias  = (const float*)d_in[4];   // [4H]
    // d_in[5] dense_w, d_in[6] dense_b: unused — softmax over size-1 axis == 1.

    // Workspace layout (512 KB total)
    char* ws = (char*)d_ws;
    unsigned short* WkT = (unsigned short*)ws;  ws += (size_t)4 * HIDDEN * EMBED  * 2;
    unsigned short* WrT = (unsigned short*)ws;  ws += (size_t)4 * HIDDEN * HIDDEN * 2;
    unsigned short* h0  = (unsigned short*)ws;  ws += (size_t)BATCH * HIDDEN * 2;
    unsigned short* h1  = (unsigned short*)ws;  ws += (size_t)BATCH * HIDDEN * 2;
    float*          cst = (float*)ws;

    const int wElems = 4 * HIDDEN * EMBED;                    // 65536
    transpose_w_bf16<<<(wElems + 255) / 256, 256, 0, stream>>>(Wk, WkT, EMBED,  4 * HIDDEN);
    transpose_w_bf16<<<(wElems + 255) / 256, 256, 0, stream>>>(Wr, WrT, HIDDEN, 4 * HIDDEN);

    const int sElems = BATCH * HIDDEN;                        // 32768
    init_state<<<(sElems + 255) / 256, 256, 0, stream>>>(h0, cst, sElems);

    dim3 grid(BATCH / 16, HIDDEN / 16);                       // 16 x 8 waves
    for (int t = 0; t < SEQ; ++t) {
        const unsigned short* hin  = (t & 1) ? h1 : h0;
        unsigned short*       hout = (t & 1) ? h0 : h1;
        lstm_step<<<grid, 32, 0, stream>>>(texts, emb, WkT, WrT, bias,
                                           hin, hout, cst, t);
    }

    write_out<<<(out_size + 255) / 256, 256, 0, stream>>>((float*)d_out, out_size);
}
